// GraphAttention_48541720379787
// MI455X (gfx1250) — matrix operand order repaired and checked by reading, not compiled, not run on hardware
//
#include <hip/hip_runtime.h>
#include <hip/hip_bf16.h>

// ---------------------------------------------------------------------------
// GAT forward for MI455X (gfx1250, wave32, WMMA + Tensor Data Mover).
// Sizes: N=4096, F=512, F_=128, H=8.
//   1) convert X, a_dst -> bf16 ; transpose+convert W -> WT bf16
//   2) WMMA GEMM: H_feat[h] = X @ W[h] (bf16 in / f32 acc), stored bf16 both
//      row-major [h][n][128] and transposed [h][128][n]
//   3) flash-attention: 4 waves/block share one head; j-tiles (32x128 + the
//      transposed 128x32 copy) are DMA'd into double-buffered LDS by the
//      Tensor Data Mover (pad_enable used for bank-conflict-free strides),
//      S = Q@H^T via WMMA, online softmax in registers, O += P@V via WMMA
//      (P C-layout -> A-layout through per-wave LDS), relu(O/l) -> out.
// a_src / att_bias are per-(h,i) constants over j -> cancel in softmax.
// ---------------------------------------------------------------------------

typedef __attribute__((ext_vector_type(16))) __bf16       v16bf;
typedef __attribute__((ext_vector_type(8)))  float        v8f;
typedef __attribute__((ext_vector_type(8)))  unsigned int v8u;
typedef __attribute__((ext_vector_type(4)))  unsigned int u32x4;
typedef __attribute__((ext_vector_type(8)))  int          i32x8;
typedef __attribute__((ext_vector_type(4)))  int          i32x4;

#define NN   4096
#define FF   512
#define FO   128
#define HH   8

#define HT_ROW   132   // 128 elem + TDM pad (2 DW per 64 DW) -> stride 66 words
#define HTT_ROW  34    // 32 elem  + TDM pad (1 DW per 16 DW) -> stride 17 words
#define NITER    (NN / 32)

#if defined(__HIP_DEVICE_COMPILE__) && __has_builtin(__builtin_amdgcn_tensor_load_to_lds)
#define GAT_HAS_TDM 1
#else
#define GAT_HAS_TDM 0
#endif

static __device__ __forceinline__ unsigned short f32_to_bf16(float f) {
  unsigned int u = __builtin_bit_cast(unsigned int, f);
  u += 0x7FFFu + ((u >> 16) & 1u);           // round-to-nearest-even
  return (unsigned short)(u >> 16);
}

// low 32 bits of a generic pointer to LDS == group-segment byte offset
static __device__ __forceinline__ unsigned lds_addr_of(const void* p) {
  return (unsigned)(uintptr_t)p;
}

// A-fragment (16x32 bf16, MxK), source row-major with given row stride.
// lane: m = lane%16 ; half = lane/16 ; VGPR v: k = 16*(v>>2) + 8*half + 2*(v&3)
static __device__ __forceinline__ v16bf
load_a_frag(const unsigned short* base, int row_stride, int lane) {
  const int m = lane & 15, half = lane >> 4;
  v8u u;
#pragma unroll
  for (int v = 0; v < 8; ++v) {
    const int kk = 16 * (v >> 2) + 8 * half + 2 * (v & 3);
    u[v] = *(const unsigned int*)(base + m * row_stride + kk);
  }
  return __builtin_bit_cast(v16bf, u);
}

// B-fragment (32x16 bf16, KxN). Memory holds B^T row-major: element B[k][n] at
// base + n*n_stride + k (k contiguous).
// lane: n = lane%16 ; half = lane/16 ; VGPR v: k = 16*(v>>2) + 8*half + 2*(v&3)
static __device__ __forceinline__ v16bf
load_b_frag(const unsigned short* base, int n_stride, int lane) {
  const int n = lane & 15, half = lane >> 4;
  v8u u;
#pragma unroll
  for (int v = 0; v < 8; ++v) {
    const int kk = 16 * (v >> 2) + 8 * half + 2 * (v & 3);
    u[v] = *(const unsigned int*)(base + n * n_stride + kk);
  }
  return __builtin_bit_cast(v16bf, u);
}

static __device__ __forceinline__ v8f wmma_bf16(v16bf a, v16bf b, v8f c) {
  return __builtin_amdgcn_wmma_f32_16x16x32_bf16(
      /*neg_a=*/false, a, /*neg_b=*/false, b,
      /*c_mod=*/(short)0, c, /*reuse_a=*/false, /*reuse_b=*/false);
}

#if GAT_HAS_TDM
// Issue one 2D TDM tile load (D# group0 + group1; groups 2/3 unused -> zero).
static __device__ __forceinline__ void
tdm_load_2d(unsigned lds_byte, unsigned long long gaddr,
            unsigned g1w0, unsigned g1w1, unsigned g1w2,
            unsigned g1w3, unsigned g1w4, unsigned g1w5) {
  u32x4 g0;
  g0[0] = 1u;                                     // count=1 (valid user D#)
  g0[1] = lds_byte;                               // lds_addr
  g0[2] = (unsigned)gaddr;                        // global_addr[31:0]
  g0[3] = (unsigned)(gaddr >> 32) | (2u << 30);   // global_addr[56:32] | type=2
  i32x8 g1;
  g1[0] = (int)g1w0; g1[1] = (int)g1w1; g1[2] = (int)g1w2; g1[3] = (int)g1w3;
  g1[4] = (int)g1w4; g1[5] = (int)g1w5; g1[6] = 0; g1[7] = 0;
  i32x4 z4 = {0, 0, 0, 0};
#if __clang_major__ >= 23
  i32x8 z8 = {0, 0, 0, 0, 0, 0, 0, 0};
  __builtin_amdgcn_tensor_load_to_lds(g0, g1, z4, z4, z8, 0);
#else
  __builtin_amdgcn_tensor_load_to_lds(g0, g1, z4, z4, 0);
#endif
}
#endif

// ------------------------------ prep kernels -------------------------------

__global__ void __launch_bounds__(256)
cvt_bf16_kernel(const float* __restrict__ src, unsigned short* __restrict__ dst, int n) {
  int i = blockIdx.x * 256 + threadIdx.x;
  if (i < n) dst[i] = f32_to_bf16(src[i]);
}

// W [H][F][F_] (f32) -> WT [H][F_][F] (bf16)
__global__ void __launch_bounds__(256)
wt_kernel(const float* __restrict__ W, unsigned short* __restrict__ WT) {
  int i = blockIdx.x * 256 + threadIdx.x;
  if (i >= HH * FF * FO) return;
  int n = i & (FO - 1);
  int f = (i >> 7) & (FF - 1);
  int h = i >> 16;
  WT[(h * FO + n) * FF + f] = f32_to_bf16(W[i]);
}

// ------------------------------ feature GEMM -------------------------------
// one wave per 16x16 tile of H_feat[h] = X(4096x512) @ W_h(512x128)

__global__ void __launch_bounds__(256)
gat_gemm_kernel(const unsigned short* __restrict__ Xb,   // [4096][512]
                const unsigned short* __restrict__ WTb,  // [H][128][512]
                unsigned short* __restrict__ Hb,         // [H][4096][128]
                unsigned short* __restrict__ HTb) {      // [H][128][4096]
  const int wid  = blockIdx.x * 8 + (threadIdx.x >> 5);  // 0..16383
  const int lane = threadIdx.x & 31;
  const int h  = wid >> 11;
  const int mt = (wid >> 3) & 255;
  const int nt = wid & 7;

  const unsigned short* Arow = Xb + mt * 16 * FF;
  const unsigned short* Brow = WTb + (h * FO + nt * 16) * FF;

  v8f C = {0.f, 0.f, 0.f, 0.f, 0.f, 0.f, 0.f, 0.f};
#pragma unroll 4
  for (int k0 = 0; k0 < FF; k0 += 32) {
    v16bf a = load_a_frag(Arow + k0, FF, lane);
    v16bf b = load_b_frag(Brow + k0, FF, lane);
    C = wmma_bf16(a, b, C);
  }

  const int n = lane & 15, half = lane >> 4;
#pragma unroll
  for (int r = 0; r < 8; ++r) {
    const int m   = mt * 16 + r + 8 * half;
    const int col = nt * 16 + n;
    const unsigned short v = f32_to_bf16(C[r]);
    Hb[(h * NN + m) * FO + col]  = v;
    HTb[(h * FO + col) * NN + m] = v;
  }
}

// --------------------------- flash attention -------------------------------
// block = 4 waves, consecutive 16-row query tiles of one head; j streamed in
// tiles of 32 via double-buffered TDM loads into LDS.

__global__ void __launch_bounds__(128)
gat_flash_kernel(const unsigned short* __restrict__ Hb,   // [H][4096][128]
                 const unsigned short* __restrict__ HTb,  // [H][128][4096]
                 const unsigned short* __restrict__ Qb,   // a_dst bf16 [H][4096][128]
                 float* __restrict__ out) {               // [4096][H*128]
  __shared__ unsigned short sHt [2][32 * HT_ROW + 8];     // row-major j-tile
  __shared__ unsigned short sHTt[2][FO * HTT_ROW + 8];    // transposed j-tile
  __shared__ unsigned short sP  [4][16 * 32];             // per-wave P staging

  const int wave = threadIdx.x >> 5;
  const int lane = threadIdx.x & 31;
  const int tile = blockIdx.x * 4 + wave;                 // 0..2047, same head/block
  const int h  = tile >> 8;
  const int i0 = (tile & 255) * 16;

  const unsigned long long Hg  = (unsigned long long)(uintptr_t)(Hb  + (size_t)h * NN * FO);
  const unsigned long long HTg = (unsigned long long)(uintptr_t)(HTb + (size_t)h * FO * NN);
  const unsigned short* Qh = Qb + ((size_t)h * NN + i0) * FO;

  // resident Q fragments: 16x128 bf16 = 4 A-frags
  v16bf qf[4];
#pragma unroll
  for (int c = 0; c < 4; ++c) qf[c] = load_a_frag(Qh + c * 32, FO, lane);

  v8f   Oacc[8];
  float mrow[8], lrow[8];
#pragma unroll
  for (int g = 0; g < 8; ++g) Oacc[g] = (v8f){0.f, 0.f, 0.f, 0.f, 0.f, 0.f, 0.f, 0.f};
#pragma unroll
  for (int r = 0; r < 8; ++r) { mrow[r] = -1e30f; lrow[r] = 0.f; }

#if GAT_HAS_TDM
  const unsigned htA[2]  = { lds_addr_of(&sHt[0][0]),  lds_addr_of(&sHt[1][0])  };
  const unsigned httA[2] = { lds_addr_of(&sHTt[0][0]), lds_addr_of(&sHTt[1][0]) };
  auto issue_tiles = [&](int buf, int j0) {
    // H_feat[h][j0..j0+31][0..127]: tile 128x32 (dim0=cols), stride0=128,
    // LDS pad: 2 DWORDs per 64 -> row stride 132 elem
    tdm_load_2d(htA[buf], Hg + (unsigned long long)j0 * (FO * 2),
                (1u << 16) | (1u << 20) | (5u << 22) | (1u << 25),
                (unsigned)FO << 16,        // tensor_dim0 = 128
                (unsigned)NN << 16,        // tensor_dim1 = 4096
                (unsigned)FO << 16,        // tile_dim0   = 128
                32u,                       // tile_dim1   = 32
                (unsigned)FO);             // tensor_dim0_stride = 128
    // HT[h][0..127][j0..j0+31]: tile 32x128 (dim0=j), stride0=4096,
    // LDS pad: 1 DWORD per 16 -> row stride 34 elem
    tdm_load_2d(httA[buf], HTg + (unsigned long long)j0 * 2,
                (1u << 16) | (1u << 20) | (3u << 22),
                (unsigned)NN << 16,        // tensor_dim0 = 4096
                (unsigned)FO << 16,        // tensor_dim1 = 128
                32u << 16,                 // tile_dim0   = 32
                (unsigned)FO,              // tile_dim1   = 128
                (unsigned)NN);             // tensor_dim0_stride = 4096
  };
  if (wave == 0) issue_tiles(0, 0);
#endif

  unsigned short* Lp = sP[wave];

  for (int it = 0; it < NITER; ++it) {
    const int j0 = it * 32;
#if GAT_HAS_TDM
    const int buf = it & 1;
    if (wave == 0) __builtin_amdgcn_s_wait_tensorcnt(0);
    __syncthreads();                       // buf ready; buf^1 free for refill
    if (wave == 0 && it + 1 < NITER) issue_tiles(buf ^ 1, j0 + 32);
#else
    const int buf = 0;
    __syncthreads();
    {  // cooperative staging fallback (plain loads)
      const unsigned short* gH  = (const unsigned short*)(uintptr_t)Hg + (size_t)j0 * FO;
      const unsigned short* gHT = (const unsigned short*)(uintptr_t)HTg + j0;
      for (int e = threadIdx.x; e < 2048; e += 128) {
        int r = e >> 6, c = (e & 63) * 2;
        *(unsigned*)&sHt[0][r * HT_ROW + c] = *(const unsigned*)&gH[r * FO + c];
      }
      for (int e = threadIdx.x; e < 2048; e += 128) {
        int r = e >> 4, c = (e & 15) * 2;
        *(unsigned*)&sHTt[0][r * HTT_ROW + c] = *(const unsigned*)&gHT[(size_t)r * NN + c];
      }
    }
    __syncthreads();
#endif
    const unsigned short* Ht  = &sHt[buf][0];
    const unsigned short* HTt = &sHTt[buf][0];

    // ---- scores S[16x32] = Q(16x128) @ H[j0..j0+31]^T (from LDS) ----
    v8f S0 = {0.f, 0.f, 0.f, 0.f, 0.f, 0.f, 0.f, 0.f};
    v8f S1 = {0.f, 0.f, 0.f, 0.f, 0.f, 0.f, 0.f, 0.f};
#pragma unroll
    for (int c = 0; c < 4; ++c) {
      v16bf b0 = load_b_frag(Ht + c * 32, HT_ROW, lane);                // cols j0..+15
      v16bf b1 = load_b_frag(Ht + 16 * HT_ROW + c * 32, HT_ROW, lane);  // cols +16..+31
      S0 = wmma_bf16(qf[c], b0, S0);
      S1 = wmma_bf16(qf[c], b1, S1);
    }

    // ---- online softmax: row max / rescale ----
    float scale[8];
#pragma unroll
    for (int r = 0; r < 8; ++r) {
      float v = fmaxf(S0[r], S1[r]);
#pragma unroll
      for (int off = 8; off >= 1; off >>= 1) v = fmaxf(v, __shfl_xor(v, off, 16));
      const float mnew = fmaxf(mrow[r], v);
      scale[r] = __expf(mrow[r] - mnew);
      mrow[r]  = mnew;
    }

    // ---- P = exp(S - m), stage to per-wave LDS, row sums ----
#pragma unroll
    for (int r = 0; r < 8; ++r) {
      const float p0 = __expf(S0[r] - mrow[r]);
      const float p1 = __expf(S1[r] - mrow[r]);
      const int m = r + 8 * (lane >> 4);
      const int n = lane & 15;
      Lp[m * 32 + n]      = f32_to_bf16(p0);
      Lp[m * 32 + 16 + n] = f32_to_bf16(p1);
      float s = p0 + p1;
#pragma unroll
      for (int off = 8; off >= 1; off >>= 1) s += __shfl_xor(s, off, 16);
      lrow[r] = lrow[r] * scale[r] + s;
    }
    asm volatile("s_wait_dscnt 0" ::: "memory");

    v16bf pf = load_a_frag(Lp, 32, lane);   // P as 16x32 A-frag

    // ---- O = O*scale + P @ V (V from transposed LDS tile) ----
#pragma unroll
    for (int g = 0; g < 8; ++g) {
#pragma unroll
      for (int r = 0; r < 8; ++r) Oacc[g][r] *= scale[r];
      v16bf bv = load_b_frag(HTt + (g * 16) * HTT_ROW, HTT_ROW, lane);
      Oacc[g] = wmma_bf16(pf, bv, Oacc[g]);
    }
  }

  // ---- epilogue: relu(O / l) -> out[n][h*128 + col] ----
  const int n = lane & 15, half = lane >> 4;
#pragma unroll
  for (int g = 0; g < 8; ++g) {
#pragma unroll
    for (int r = 0; r < 8; ++r) {
      const int m = i0 + r + 8 * half;
      const float v = Oacc[g][r] / lrow[r];
      out[(size_t)m * (HH * FO) + h * FO + g * 16 + n] = fmaxf(v, 0.f);
    }
  }
}

// ------------------------------- launcher ----------------------------------

extern "C" void kernel_launch(void* const* d_in, const int* in_sizes, int n_in,
                              void* d_out, int out_size, void* d_ws, size_t ws_size,
                              hipStream_t stream) {
  (void)in_sizes; (void)n_in; (void)out_size; (void)ws_size;

  const float* X    = (const float*)d_in[0];  // [4096][512]
  const float* W    = (const float*)d_in[1];  // [8][512][128]
  // d_in[2] (a_src) and d_in[4] (att_bias) are row-constant softmax shifts: unused.
  const float* Adst = (const float*)d_in[3];  // [8][4096][128]
  float* out = (float*)d_out;                 // [4096][1024]

  char* ws = (char*)d_ws;
  unsigned short* Xb  = (unsigned short*)ws;  ws += (size_t)NN * FF * 2;        // 4 MB
  unsigned short* WTb = (unsigned short*)ws;  ws += (size_t)HH * FO * FF * 2;   // 1 MB
  unsigned short* Qb  = (unsigned short*)ws;  ws += (size_t)HH * NN * FO * 2;   // 8 MB
  unsigned short* Hb  = (unsigned short*)ws;  ws += (size_t)HH * NN * FO * 2;   // 8 MB
  unsigned short* HTb = (unsigned short*)ws;                                    // 8 MB

  cvt_bf16_kernel<<<(NN * FF + 255) / 256, 256, 0, stream>>>(X, Xb, NN * FF);
  wt_kernel<<<(HH * FF * FO + 255) / 256, 256, 0, stream>>>(W, WTb);
  cvt_bf16_kernel<<<(HH * NN * FO + 255) / 256, 256, 0, stream>>>(Adst, Qb, HH * NN * FO);

  // 16384 waves = H * (4096/16) * (128/16), 8 waves / 256-thread block
  gat_gemm_kernel<<<2048, 256, 0, stream>>>(Xb, WTb, Hb, HTb);

  // 2048 waves = H * (4096/16), 4 waves / 128-thread block
  gat_flash_kernel<<<512, 128, 0, stream>>>(Hb, HTb, Qb, out);
}
